// MambaBlock_120259085012
// MI455X (gfx1250) — matrix operand order repaired and checked by reading, not compiled, not run on hardware
//
#include <hip/hip_runtime.h>
#include <stdint.h>
#include <stddef.h>

// ---- problem constants (from reference) ----
#define NTOK   4096   // B*S = 2*2048
#define SEQ    2048
#define DMODEL 1024
#define DINNER 2048
#define DSTATE 16
#define EPSR   1.1920928955078125e-07f  // float32 eps

// ---- CDNA5 WMMA types ----
typedef __attribute__((ext_vector_type(16))) __bf16 v16bf;
typedef __attribute__((ext_vector_type(8)))  float  v8f;

struct alignas(16) U128 { uint32_t a, b, c, d; };
union FragBF { v16bf v; U128 q[2]; };

__device__ __forceinline__ uint16_t f2bf(float f) {
  union { float f; uint32_t u; } cv; cv.f = f;
  uint32_t u = cv.u;
  uint32_t r = u + 0x7FFFu + ((u >> 16) & 1u);   // round-to-nearest-even
  if ((u & 0x7F800000u) == 0x7F800000u) r = u;   // Inf/NaN passthrough
  return (uint16_t)(r >> 16);
}

__device__ __forceinline__ float sigm(float v) { return 1.0f / (1.0f + __expf(-v)); }

// ---------------- f32 -> bf16 convert ----------------
__global__ void cvt_f32_bf16(const float* __restrict__ in, uint16_t* __restrict__ out, int n) {
  int i = blockIdx.x * blockDim.x + threadIdx.x;
  int stride = gridDim.x * blockDim.x;
  for (; i < n; i += stride) out[i] = f2bf(in[i]);
}

// ---------------- WMMA GEMM: C[M,N] = A[M,K](bf16) * W[N,K](bf16)^T ----------------
// block = 256 threads = 8 waves arranged 2(M) x 4(N); block tile 64 x 256.
// wave tile 32(M) x 64(N): 2 A-fragments x 4 B-fragments -> 8 WMMAs per k-step.
// Requires M%64==0, N%256==0, K%32==0.
__global__ __launch_bounds__(256) void gemm_bf16_wmma(
    const uint16_t* __restrict__ A,   // M x K row-major
    const uint16_t* __restrict__ W,   // N x K row-major
    float* __restrict__ C,            // M x N row-major
    int M, int N, int K) {
  const int tid   = threadIdx.x;
  const int wave  = tid >> 5;
  const int lane  = tid & 31;        // wave32
  const int lrow  = lane & 15;
  const int lhalf = lane >> 4;

  const int m0 = blockIdx.y * 64 + (wave >> 2) * 32;   // wave owns rows [m0, m0+32)
  const int n0 = blockIdx.x * 256 + (wave & 3) * 64;   // wave owns cols [n0, n0+64)

  // A fragment (16x32 bf16, MxK): lane<16 -> row=lrow, K={kk..kk+7, kk+16..kk+23}
  //                               lane>=16 -> K={kk+8..kk+15, kk+24..kk+31}
  const uint16_t* __restrict__ Arow = A + (size_t)(m0 + lrow) * K;
  // B fragment (32x16 bf16, KxN): lane = column; lanes 0-15 hold K={kk..kk+7, kk+16..kk+23},
  //                               lanes 16-31 hold K={kk+8..kk+15, kk+24..kk+31}
  const uint16_t* __restrict__ Wrow = W + (size_t)(n0 + lrow) * K;

  const int aoff0 = lhalf * 8;
  const int aoff1 = 16 + lhalf * 8;
  const int boff  = lhalf * 8;
  const int AR1   = 16 * K;          // +16 rows of A (elements)

  v8f acc[2][4];
  #pragma unroll
  for (int i = 0; i < 2; ++i)
    #pragma unroll
    for (int j = 0; j < 4; ++j)
      acc[i][j] = (v8f){0.f, 0.f, 0.f, 0.f, 0.f, 0.f, 0.f, 0.f};

  for (int kk = 0; kk < K; kk += 32) {
    FragBF fa0, fa1;
    fa0.q[0] = *(const U128*)(Arow + kk + aoff0);
    fa0.q[1] = *(const U128*)(Arow + kk + aoff1);
    fa1.q[0] = *(const U128*)(Arow + AR1 + kk + aoff0);
    fa1.q[1] = *(const U128*)(Arow + AR1 + kk + aoff1);
    #pragma unroll
    for (int j = 0; j < 4; ++j) {
      FragBF fb;
      const uint16_t* wp = Wrow + j * 16 * K + kk + boff;  // uniform col-tile offset
      fb.q[0] = *(const U128*)(wp);
      fb.q[1] = *(const U128*)(wp + 16);
      acc[0][j] = __builtin_amdgcn_wmma_f32_16x16x32_bf16(
          false, fa0.v, false, fb.v, (short)0, acc[0][j], false, false);
      acc[1][j] = __builtin_amdgcn_wmma_f32_16x16x32_bf16(
          false, fa1.v, false, fb.v, (short)0, acc[1][j], false, false);
    }
  }

  // C layout: VGPR v -> M = mtile + v + 8*lhalf, N = ntile + lrow
  #pragma unroll
  for (int i = 0; i < 2; ++i) {
    #pragma unroll
    for (int j = 0; j < 4; ++j) {
      float* __restrict__ Cc =
          C + (size_t)(m0 + 16 * i + lhalf * 8) * N + (n0 + 16 * j + lrow);
      #pragma unroll
      for (int v = 0; v < 8; ++v) Cc[(size_t)v * N] = acc[i][j][v];
    }
  }
}

// ---------------- depthwise causal conv1d (k=4) + SiLU ----------------
// reads x_in = xz[:, 0:DINNER]; writes xc in f32 and bf16
__global__ void conv_silu_kernel(const float* __restrict__ xz,
                                 const float* __restrict__ cw,   // (DINNER,1,4)
                                 const float* __restrict__ cb,   // (DINNER,)
                                 float* __restrict__ xcf,
                                 uint16_t* __restrict__ xcb) {
  int i = blockIdx.x * blockDim.x + threadIdx.x;
  if (i >= NTOK * DINNER) return;
  int m = i / DINNER, c = i - m * DINNER;
  int s = m & (SEQ - 1);
  const ptrdiff_t rs = 2 * DINNER;               // xz row stride
  const float* col = xz + (size_t)m * rs + c;    // x_in column
  float w0 = cw[c * 4 + 0], w1 = cw[c * 4 + 1], w2 = cw[c * 4 + 2], w3 = cw[c * 4 + 3];
  float acc = cb[c] + w3 * col[0];
  if (s >= 1) acc += w2 * col[-rs];
  if (s >= 2) acc += w1 * col[-2 * rs];
  if (s >= 3) acc += w0 * col[-3 * rs];
  float v = acc * sigm(acc);
  xcf[i] = v;
  xcb[i] = f2bf(v);
}

// ---------------- B_mat = xc @ W_x[0:16].T (tiny N) ----------------
__global__ void bmat_kernel(const float* __restrict__ xc,
                            const float* __restrict__ Wx,   // (32, DINNER); rows 0..15 = B
                            float* __restrict__ Bm) {       // (NTOK, 16)
  int idx = blockIdx.x * blockDim.x + threadIdx.x;
  if (idx >= NTOK * DSTATE) return;
  int m = idx >> 4, n = idx & 15;
  const float* a = xc + (size_t)m * DINNER;
  const float* w = Wx + (size_t)n * DINNER;
  float acc = 0.f;
  for (int k = 0; k < DINNER; ++k) acc += a[k] * w[k];
  Bm[idx] = acc;
}

// ---------------- selective elementwise: softplus, state sum, gates ----------------
__global__ void selective_kernel(const float* __restrict__ dtpre,
                                 const float* __restrict__ bdt,
                                 const float* __restrict__ xcf,
                                 const float* __restrict__ Alog,   // (DINNER,16)
                                 const float* __restrict__ Bm,     // (NTOK,16)
                                 const float* __restrict__ Dp,     // (DINNER,)
                                 const float* __restrict__ xz,     // z = cols [DINNER:2*DINNER)
                                 uint16_t* __restrict__ yin) {
  int i = blockIdx.x * blockDim.x + threadIdx.x;
  if (i >= NTOK * DINNER) return;
  int m = i / DINNER, c = i - m * DINNER;
  float p  = dtpre[i] + bdt[c];
  float dt = (p > 20.f) ? p : log1pf(__expf(p));      // softplus
  float xv = xcf[i];
  float acc = 0.f;
  #pragma unroll
  for (int n = 0; n < DSTATE; ++n) {
    float A = -__expf(Alog[c * DSTATE + n]);
    acc += __expf(dt * A) * Bm[m * DSTATE + n];
  }
  float y = acc * dt * xv + Dp[c] * xv;
  float z = xz[(size_t)m * (2 * DINNER) + DINNER + c];
  y *= z * sigm(z);
  yin[i] = f2bf(y);
}

// ---------------- residual add + RMSNorm (one block per token row) ----------------
__global__ __launch_bounds__(256) void addnorm_kernel(const float* __restrict__ ypre,
                                                      const float* __restrict__ x,
                                                      const float* __restrict__ nw,
                                                      float* __restrict__ out) {
  __shared__ float red[256];
  int m = blockIdx.x, tid = threadIdx.x;
  const float* yr = ypre + (size_t)m * DMODEL;
  const float* xr = x    + (size_t)m * DMODEL;
  float t[4]; float s = 0.f;
  #pragma unroll
  for (int j = 0; j < 4; ++j) {
    int c = tid + j * 256;
    t[j] = yr[c] + xr[c];
    s += t[j] * t[j];
  }
  red[tid] = s;
  __syncthreads();
  for (int w = 128; w > 0; w >>= 1) {
    if (tid < w) red[tid] += red[tid + w];
    __syncthreads();
  }
  float inv = rsqrtf(red[0] * (1.0f / DMODEL) + EPSR);
  float* orow = out + (size_t)m * DMODEL;
  #pragma unroll
  for (int j = 0; j < 4; ++j) {
    int c = tid + j * 256;
    orow[c] = t[j] * inv * nw[c];
  }
}

// ---------------- launcher ----------------
extern "C" void kernel_launch(void* const* d_in, const int* in_sizes, int n_in,
                              void* d_out, int out_size, void* d_ws, size_t ws_size,
                              hipStream_t stream) {
  const float* x      = (const float*)d_in[0];
  const float* W_in   = (const float*)d_in[1];
  const float* conv_w = (const float*)d_in[2];
  const float* conv_b = (const float*)d_in[3];
  const float* W_x    = (const float*)d_in[4];
  const float* W_dt   = (const float*)d_in[5];
  const float* b_dt   = (const float*)d_in[6];
  const float* A_log  = (const float*)d_in[7];
  const float* Dp     = (const float*)d_in[8];
  const float* W_out  = (const float*)d_in[9];
  const float* norm_w = (const float*)d_in[10];
  float* out = (float*)d_out;

  // carve workspace (offsets 256B-aligned)
  size_t off = 0;
  char* base = (char*)d_ws;
  auto take = [&](size_t bytes) -> void* {
    void* p = base + off;
    off += (bytes + 255) & ~(size_t)255;
    return p;
  };
  uint16_t* x_bf    = (uint16_t*)take((size_t)NTOK * DMODEL * 2);
  uint16_t* Win_bf  = (uint16_t*)take((size_t)2 * DINNER * DMODEL * 2);
  uint16_t* Wdt_bf  = (uint16_t*)take((size_t)DINNER * DINNER * 2);
  uint16_t* Wout_bf = (uint16_t*)take((size_t)DMODEL * DINNER * 2);
  float*    xz_f    = (float*)take((size_t)NTOK * 2 * DINNER * 4);
  float*    xc_f    = (float*)take((size_t)NTOK * DINNER * 4);
  uint16_t* xc_bf   = (uint16_t*)take((size_t)NTOK * DINNER * 2);
  float*    dtpre_f = (float*)take((size_t)NTOK * DINNER * 4);
  float*    bmat_f  = (float*)take((size_t)NTOK * DSTATE * 4);
  uint16_t* yin_bf  = (uint16_t*)take((size_t)NTOK * DINNER * 2);
  float*    ypre_f  = (float*)take((size_t)NTOK * DMODEL * 4);
  (void)ws_size; (void)in_sizes; (void)n_in; (void)out_size;

  const int TPB = 256;
  // 1) bf16 conversions
  {
    int n;
    n = NTOK * DMODEL;       cvt_f32_bf16<<<(n + TPB - 1) / TPB, TPB, 0, stream>>>(x,     x_bf,    n);
    n = 2 * DINNER * DMODEL; cvt_f32_bf16<<<(n + TPB - 1) / TPB, TPB, 0, stream>>>(W_in,  Win_bf,  n);
    n = DINNER * DINNER;     cvt_f32_bf16<<<(n + TPB - 1) / TPB, TPB, 0, stream>>>(W_dt,  Wdt_bf,  n);
    n = DMODEL * DINNER;     cvt_f32_bf16<<<(n + TPB - 1) / TPB, TPB, 0, stream>>>(W_out, Wout_bf, n);
  }
  // 2) xz = x @ W_in^T   (M=4096, N=4096, K=1024)
  gemm_bf16_wmma<<<dim3((2 * DINNER) / 256, NTOK / 64), TPB, 0, stream>>>(
      x_bf, Win_bf, xz_f, NTOK, 2 * DINNER, DMODEL);
  // 3) depthwise causal conv + SiLU
  conv_silu_kernel<<<(NTOK * DINNER) / TPB, TPB, 0, stream>>>(xz_f, conv_w, conv_b, xc_f, xc_bf);
  // 4) dt_pre = xc @ W_dt^T  (M=4096, N=2048, K=2048)
  gemm_bf16_wmma<<<dim3(DINNER / 256, NTOK / 64), TPB, 0, stream>>>(
      xc_bf, Wdt_bf, dtpre_f, NTOK, DINNER, DINNER);
  // 5) B_mat = xc @ W_x[0:16]^T
  bmat_kernel<<<(NTOK * DSTATE) / TPB, TPB, 0, stream>>>(xc_f, W_x, bmat_f);
  // 6) selective elementwise -> y_inner (bf16)
  selective_kernel<<<(NTOK * DINNER) / TPB, TPB, 0, stream>>>(
      dtpre_f, b_dt, xc_f, A_log, bmat_f, Dp, xz_f, yin_bf);
  // 7) y_pre = y_inner @ W_out^T  (M=4096, N=1024, K=2048)
  gemm_bf16_wmma<<<dim3(DMODEL / 256, NTOK / 64), TPB, 0, stream>>>(
      yin_bf, Wout_bf, ypre_f, NTOK, DMODEL, DINNER);
  // 8) residual + RMSNorm -> out
  addnorm_kernel<<<NTOK, TPB, 0, stream>>>(ypre_f, x, norm_w, out);
}
